// UPT_28132035789196
// MI455X (gfx1250) — matrix operand order repaired and checked by reading, not compiled, not run on hardware
//
#include <hip/hip_runtime.h>
#include <hip/hip_bf16.h>
#include <stdint.h>

// ---------------------------------------------------------------------------
// UPT supernode message-passing, restructured for MI455X (gfx1250, wave32):
//   out = [ mean_mask(gelu(src@W1a + sn@W1b + b1)) || sn ] @ [[W2@Wp_a],[Wp_b]]
//         + (b2@Wp_a + b_proj)
// All GEMMs via v_wmma_f32_16x16x32_bf16 (bf16 in, f32 accum).
// ---------------------------------------------------------------------------

#define HIDDEN  384
#define NPOINTS 100000
#define NSUPER  2048
#define MAXDEG  32
#define NT      24              // 384 / 16 N-tiles

typedef __attribute__((ext_vector_type(16))) __bf16 v16bf;
typedef __attribute__((ext_vector_type(8)))  float  v8f;

union ABFrag { uint32_t u[8]; v16bf v; };

__device__ __forceinline__ uint16_t f2bf(float f) {
  uint32_t u = __float_as_uint(f);
  u += 0x7fffu + ((u >> 16) & 1u);          // round-to-nearest-even
  return (uint16_t)(u >> 16);
}
__device__ __forceinline__ float geluf(float y) {
  return 0.5f * y * (1.0f + erff(y * 0.70710678118654752440f));
}

// ---------------------------------------------------------------------------
// K1: Wfold = W2 @ W_proj[0:384], bfold = b2 @ W_proj[0:384] + b_proj  (f32)
// ---------------------------------------------------------------------------
__global__ void prep_fold(const float* __restrict__ W2, const float* __restrict__ Wp,
                          const float* __restrict__ b2, const float* __restrict__ bp,
                          float* __restrict__ Wfold, float* __restrict__ bfold) {
  int id = blockIdx.x * blockDim.x + threadIdx.x;
  if (id < HIDDEN * HIDDEN) {
    int k = id / HIDDEN, n = id % HIDDEN;
    float acc = 0.f;
    for (int j = 0; j < HIDDEN; ++j) acc += W2[k * HIDDEN + j] * Wp[j * HIDDEN + n];
    Wfold[id] = acc;
  } else if (id < HIDDEN * HIDDEN + HIDDEN) {
    int n = id - HIDDEN * HIDDEN;
    float acc = bp[n];
    for (int j = 0; j < HIDDEN; ++j) acc += b2[j] * Wp[j * HIDDEN + n];
    bfold[n] = acc;
  }
}

// ---------------------------------------------------------------------------
// K2: pack a (K x 384) f32 weight into WMMA-B fragment-ready bf16 layout.
// Per ISA 05_wmma: B 32x16 bf16, lane l holds column n = l%16,
// K rows (l<16 ? 0..7,16..23 : 8..15,24..31) packed 2-per-dword (low = even k).
// Storage: dst[((kt*NT + nt)*32 + lane)*8 + j]  -> 32 contiguous B/lane.
// Rows k < split come from s0, rows >= split from s1 (both row-major, ld=384).
// ---------------------------------------------------------------------------
__global__ void prep_frag(const float* __restrict__ s0, const float* __restrict__ s1,
                          int split, int KT, uint32_t* __restrict__ dst) {
  int id = blockIdx.x * blockDim.x + threadIdx.x;
  int total = KT * NT * 32 * 8;
  if (id >= total) return;
  int j    = id & 7;
  int lane = (id >> 3) & 31;
  int nt   = (id >> 8) % NT;
  int kt   = id / (NT * 32 * 8);
  int n    = nt * 16 + (lane & 15);
  int k    = kt * 32 + ((j >> 2) << 4) + ((lane >> 4) << 3) + ((j & 3) << 1);
  float a = (k     < split) ? s0[(size_t)k * HIDDEN + n]       : s1[(size_t)(k - split) * HIDDEN + n];
  float b = (k + 1 < split) ? s0[(size_t)(k + 1) * HIDDEN + n] : s1[(size_t)(k + 1 - split) * HIDDEN + n];
  dst[id] = (uint32_t)f2bf(a) | ((uint32_t)f2bf(b) << 16);
}

// ---------------------------------------------------------------------------
// K3: x = sincos_embed(pos) + feat @ W_in + b_in   -> bf16 (100000 x 384)
// emb layout: c = d*128 + w ; w<64: sin(pos_d * f_w), else cos(pos_d * f_{w-64})
// ---------------------------------------------------------------------------
__global__ void embed(const float* __restrict__ pos, const float* __restrict__ feat,
                      const float* __restrict__ Win, const float* __restrict__ bin,
                      uint16_t* __restrict__ x) {
  int i = blockIdx.x;
  __shared__ float p[3], f[3];
  if (threadIdx.x < 3) { p[threadIdx.x] = pos[i * 3 + threadIdx.x];
                         f[threadIdx.x] = feat[i * 3 + threadIdx.x]; }
  __syncthreads();
  for (int c = threadIdx.x; c < HIDDEN; c += blockDim.x) {
    int d = c >> 7, w = c & 127, jj = w & 63;
    float fr  = __expf(-(float)jj * (9.210340371976184f / 64.0f));   // 10000^(-j/64)
    float ang = p[d] * fr;
    float e   = (w < 64) ? sinf(ang) : cosf(ang);
    float val = e + f[0] * Win[c] + f[1] * Win[HIDDEN + c] + f[2] * Win[2 * HIDDEN + c] + bin[c];
    x[(size_t)i * HIDDEN + c] = f2bf(val);
  }
}

// ---------------------------------------------------------------------------
// K4: gather supernode rows into the right half of A2 (2048 x 768 bf16)
// ---------------------------------------------------------------------------
__global__ void gather_sn(const uint16_t* __restrict__ x, const int* __restrict__ snidx,
                          uint16_t* __restrict__ A2) {
  int s = blockIdx.x;
  size_t src = (size_t)snidx[s] * HIDDEN;
  for (int c = threadIdx.x; c < HIDDEN; c += blockDim.x)
    A2[(size_t)s * 768 + 384 + c] = x[src + c];
}

// ---------------------------------------------------------------------------
// K5 (x2): generic WMMA GEMM: out(Mx384,f32) = A(MxK,bf16) @ Bfrag + bias.
// One m-tile (16 rows) per block; 128 threads = 4 waves; 6 n-tiles per wave.
// A tile staged in LDS (stride K+8 to spread banks), read per ISA A layout:
// lane l -> row l%16; dword j -> k = kt*32 + (j>>2)*16 + (l>>4)*8 + (j&3)*2.
// ---------------------------------------------------------------------------
__global__ __launch_bounds__(128) void gemm_frag(
    const uint16_t* __restrict__ A, int lda, const uint32_t* __restrict__ Bfrag,
    const float* __restrict__ bias, float* __restrict__ out, int ldc, int K) {
  __shared__ __align__(16) uint16_t As[16 * (768 + 8)];
  const int sA = K + 8;
  const int mtile = blockIdx.x;
  const int t = threadIdx.x;
  const int cpr = K >> 3;                        // 16B chunks per row
  for (int chunk = t; chunk < 16 * cpr; chunk += 128) {
    int row = chunk / cpr, off = (chunk - row * cpr) << 3;
    *(uint4*)(&As[row * sA + off]) =
        *(const uint4*)(A + (size_t)(mtile * 16 + row) * lda + off);
  }
  __syncthreads();
  const int wave = t >> 5, lane = t & 31;
  const int rowA  = lane & 15;
  const int khalf = (lane >> 4) << 3;
  const int KT = K >> 5;
  for (int nti = 0; nti < 6; ++nti) {
    const int nt = wave * 6 + nti;
    v8f acc = {0.f, 0.f, 0.f, 0.f, 0.f, 0.f, 0.f, 0.f};
    for (int kt = 0; kt < KT; ++kt) {
      ABFrag a, b;
      const uint16_t* ap = &As[rowA * sA + kt * 32];
#pragma unroll
      for (int j = 0; j < 8; ++j) {
        int kloc = ((j >> 2) << 4) + khalf + ((j & 3) << 1);
        a.u[j] = *(const uint32_t*)(ap + kloc);
      }
      const uint4* bp = (const uint4*)(Bfrag + (((size_t)kt * NT + nt) * 32 + lane) * 8);
      uint4 blo = bp[0], bhi = bp[1];
      b.u[0] = blo.x; b.u[1] = blo.y; b.u[2] = blo.z; b.u[3] = blo.w;
      b.u[4] = bhi.x; b.u[5] = bhi.y; b.u[6] = bhi.z; b.u[7] = bhi.w;
      acc = __builtin_amdgcn_wmma_f32_16x16x32_bf16(false, a.v, false, b.v,
                                                    (short)0, acc, false, false);
    }
    const int col = nt * 16 + (lane & 15);
    const float bb = bias[col];
    const int r0 = (lane >> 4) << 3;             // C/D: lanes>=16 hold rows 8..15
#pragma unroll
    for (int r = 0; r < 8; ++r)
      out[(size_t)(mtile * 16 + r0 + r) * ldc + col] = acc[r] + bb;
  }
}

// ---------------------------------------------------------------------------
// K6: per-supernode message passing. One block (4 waves) per supernode.
// Y(32x384) = src @ W1a (WMMA) + V[s] ; gbar = mean_mask(gelu(Y)) -> bf16
// into A2 columns 0..383. Row reduction uses the C-layout + shfl_xor(16).
// ---------------------------------------------------------------------------
__global__ __launch_bounds__(128) void mpnn(
    const uint16_t* __restrict__ x, const int* __restrict__ nbr,
    const unsigned char* __restrict__ mask, const float* __restrict__ V,
    const uint32_t* __restrict__ W1aFrag, uint16_t* __restrict__ A2) {
  __shared__ __align__(16) uint16_t As[32 * 392];
  __shared__ float maskf[32];
  __shared__ int   nb[32];
  __shared__ float invc;
  const int s = blockIdx.x, t = threadIdx.x;
  if (t < 32) {
    nb[t]    = nbr[s * 32 + t];
    maskf[t] = mask[s * 32 + t] ? 1.0f : 0.0f;
  }
  __syncthreads();
  if (t == 0) {
    float c = 0.f;
    for (int i = 0; i < 32; ++i) c += maskf[i];
    invc = 1.0f / fmaxf(c, 1.0f);
  }
  for (int chunk = t; chunk < 32 * 48; chunk += 128) {  // 32 rows x 768B
    int row = chunk / 48, off = (chunk - row * 48) << 3;
    *(uint4*)(&As[row * 392 + off]) =
        *(const uint4*)(x + (size_t)nb[row] * HIDDEN + off);
  }
  __syncthreads();
  const int wave = t >> 5, lane = t & 31;
  const int khalf = (lane >> 4) << 3;
  for (int nti = 0; nti < 6; ++nti) {
    const int nt = wave * 6 + nti;
    v8f acc0 = {0.f, 0.f, 0.f, 0.f, 0.f, 0.f, 0.f, 0.f};
    v8f acc1 = {0.f, 0.f, 0.f, 0.f, 0.f, 0.f, 0.f, 0.f};
    for (int kt = 0; kt < 12; ++kt) {
      ABFrag a0, a1, b;
      const uint4* bp = (const uint4*)(W1aFrag + (((size_t)kt * NT + nt) * 32 + lane) * 8);
      uint4 blo = bp[0], bhi = bp[1];
      b.u[0] = blo.x; b.u[1] = blo.y; b.u[2] = blo.z; b.u[3] = blo.w;
      b.u[4] = bhi.x; b.u[5] = bhi.y; b.u[6] = bhi.z; b.u[7] = bhi.w;
      const uint16_t* ap0 = &As[(lane & 15) * 392 + kt * 32];          // rows 0..15
      const uint16_t* ap1 = &As[((lane & 15) + 16) * 392 + kt * 32];   // rows 16..31
#pragma unroll
      for (int j = 0; j < 8; ++j) {
        int kloc = ((j >> 2) << 4) + khalf + ((j & 3) << 1);
        a0.u[j] = *(const uint32_t*)(ap0 + kloc);
        a1.u[j] = *(const uint32_t*)(ap1 + kloc);
      }
      acc0 = __builtin_amdgcn_wmma_f32_16x16x32_bf16(false, a0.v, false, b.v,
                                                     (short)0, acc0, false, false);
      acc1 = __builtin_amdgcn_wmma_f32_16x16x32_bf16(false, a1.v, false, b.v,
                                                     (short)0, acc1, false, false);
    }
    const int col = nt * 16 + (lane & 15);
    const float v = V[(size_t)s * HIDDEN + col];   // sn@W1b + b1
    const int r0 = (lane >> 4) << 3;
    float partial = 0.f;
#pragma unroll
    for (int r = 0; r < 8; ++r) {
      int row0 = r0 + r;
      partial += geluf(acc0[r] + v) * maskf[row0];
      partial += geluf(acc1[r] + v) * maskf[16 + row0];
    }
    float tot = partial + __shfl_xor(partial, 16, 32);  // combine lane halves
    if (lane < 16) A2[(size_t)s * 768 + col] = f2bf(tot * invc);
  }
}

// ---------------------------------------------------------------------------
extern "C" void kernel_launch(void* const* d_in, const int* in_sizes, int n_in,
                              void* d_out, int out_size, void* d_ws, size_t ws_size,
                              hipStream_t stream) {
  (void)in_sizes; (void)n_in; (void)out_size; (void)ws_size;
  const float* pos  = (const float*)d_in[0];
  const float* feat = (const float*)d_in[1];
  const int*   sni  = (const int*)d_in[2];
  const int*   nbr  = (const int*)d_in[3];
  const unsigned char* msk = (const unsigned char*)d_in[4];
  const float* Win  = (const float*)d_in[5];
  const float* bin  = (const float*)d_in[6];
  const float* W1   = (const float*)d_in[7];
  const float* b1   = (const float*)d_in[8];
  const float* W2   = (const float*)d_in[9];
  const float* b2   = (const float*)d_in[10];
  const float* Wp   = (const float*)d_in[11];
  const float* bp   = (const float*)d_in[12];
  float* out = (float*)d_out;

  uint8_t* ws = (uint8_t*)d_ws;
  size_t off = 0;
  auto alloc = [&](size_t bytes) -> void* {
    void* p = ws + off;
    off = (off + bytes + 255) & ~(size_t)255;
    return p;
  };
  uint16_t* x     = (uint16_t*)alloc((size_t)NPOINTS * HIDDEN * 2);  // 76.8 MB
  uint16_t* A2    = (uint16_t*)alloc((size_t)NSUPER * 768 * 2);      // [gbar || sn]
  float*    V     = (float*)   alloc((size_t)NSUPER * HIDDEN * 4);
  uint32_t* W1aF  = (uint32_t*)alloc((size_t)12 * NT * 32 * 32);
  uint32_t* W1bF  = (uint32_t*)alloc((size_t)12 * NT * 32 * 32);
  uint32_t* WcatF = (uint32_t*)alloc((size_t)24 * NT * 32 * 32);
  float*    Wfold = (float*)   alloc((size_t)HIDDEN * HIDDEN * 4);
  float*    bfold = (float*)   alloc((size_t)HIDDEN * 4);

  prep_fold<<<(HIDDEN * HIDDEN + HIDDEN + 255) / 256, 256, 0, stream>>>(W2, Wp, b2, bp, Wfold, bfold);
  prep_frag<<<(12 * NT * 32 * 8 + 255) / 256, 256, 0, stream>>>(W1, W1, 1 << 30, 12, W1aF);
  prep_frag<<<(12 * NT * 32 * 8 + 255) / 256, 256, 0, stream>>>(W1 + HIDDEN * HIDDEN,
                                                                W1 + HIDDEN * HIDDEN, 1 << 30, 12, W1bF);
  prep_frag<<<(24 * NT * 32 * 8 + 255) / 256, 256, 0, stream>>>(Wfold, Wp + HIDDEN * HIDDEN,
                                                                HIDDEN, 24, WcatF);
  embed<<<NPOINTS, 128, 0, stream>>>(pos, feat, Win, bin, x);
  gather_sn<<<NSUPER, 128, 0, stream>>>(x, sni, A2);
  // V = SN @ W1b + b1   (A = sn half of A2, lda=768)
  gemm_frag<<<NSUPER / 16, 128, 0, stream>>>(A2 + 384, 768, W1bF, b1, V, HIDDEN, HIDDEN);
  mpnn<<<NSUPER, 128, 0, stream>>>(x, nbr, msk, V, W1aF, A2);
  // out = [gbar || sn] @ [[Wfold],[Wp_b]] + bfold
  gemm_frag<<<NSUPER / 16, 128, 0, stream>>>(A2, 768, WcatF, bfold, out, HIDDEN, 768);
}
